// BiMambaLayer_28106265985583
// MI455X (gfx1250) — matrix operand order repaired and checked by reading, not compiled, not run on hardware
//
#include <hip/hip_runtime.h>
#include <hip/hip_bf16.h>
#include <math.h>

// ---------------- problem constants (from reference) ----------------
#define B_   32
#define L_   2048
#define D_   128
#define DI_  256          // D_INNER
#define DS_  16           // D_STATE
#define DTR_ 8            // DT_RANK
#define XPN_ 40           // DT_RANK + 2*D_STATE
#define MROWS (B_ * L_)   // 65536

typedef __attribute__((ext_vector_type(16))) __bf16 v16bf;
typedef __attribute__((ext_vector_type(8)))  float  v8f;

static __device__ __forceinline__ float sigmoidf_(float v) {
    return 1.0f / (1.0f + __expf(-v));
}

// =====================================================================
// GEMM: C[M,N] = A[M,K] * W[N,K]^T (+bias).  A,W pre-converted to bf16;
// fragments are direct b128 loads already in WMMA register order.
// Each wave: 32x32 macro-tile (2 M-tiles x 2 N-tiles, 4 accumulators),
// 4 waves/block along M.  Layouts per CDNA5 ISA 7.12.2:
//   A lane fragment = two contiguous 8-elem K runs (k0+half*8, k0+16+half*8)
//   B lane fragment = two contiguous 8-elem K runs (k0+half*8, k0+16+half*8)
//   C/D: VGPR r -> row r + half*8, col = l16.
// =====================================================================
template<int N, int K>
__global__ void gemm_wmma_kernel(const __bf16* __restrict__ A,
                                 const __bf16* __restrict__ W,   // [N,K] bf16
                                 const float* __restrict__ bias,
                                 float* __restrict__ C)
{
    constexpr int KSTEPS = K / 32;
    const int wave = threadIdx.x >> 5;
    const int lane = threadIdx.x & 31;
    const int half = lane >> 4;
    const int l16  = lane & 15;

    const int m0 = (blockIdx.x * 4 + wave) * 32;
    const int n0 = blockIdx.y * 32;

    const long arow0 = (long)(m0 + l16) * K;
    const long arow1 = arow0 + (long)16 * K;
    int  ncol[2];
    long wrow[2];
#pragma unroll
    for (int t = 0; t < 2; ++t) {
        ncol[t] = n0 + t * 16 + l16;
        const int ncl = (N % 32) ? ((ncol[t] < N) ? ncol[t] : (N - 1)) : ncol[t];
        wrow[t] = (long)ncl * K;
    }

    union frag { uint4 q[2]; v16bf v; };
    v8f acc00 = {0.f,0.f,0.f,0.f,0.f,0.f,0.f,0.f};
    v8f acc01 = acc00, acc10 = acc00, acc11 = acc00;

#pragma unroll
    for (int ks = 0; ks < KSTEPS; ++ks) {
        const int k0 = ks * 32;
        frag a0, a1, b0, b1;
        a0.q[0] = *(const uint4*)(A + arow0 + k0 + half * 8);
        a0.q[1] = *(const uint4*)(A + arow0 + k0 + 16 + half * 8);
        a1.q[0] = *(const uint4*)(A + arow1 + k0 + half * 8);
        a1.q[1] = *(const uint4*)(A + arow1 + k0 + 16 + half * 8);
        b0.q[0] = *(const uint4*)(W + wrow[0] + k0 + half * 8);
        b0.q[1] = *(const uint4*)(W + wrow[0] + k0 + 16 + half * 8);
        b1.q[0] = *(const uint4*)(W + wrow[1] + k0 + half * 8);
        b1.q[1] = *(const uint4*)(W + wrow[1] + k0 + 16 + half * 8);
        acc00 = __builtin_amdgcn_wmma_f32_16x16x32_bf16(false, a0.v, false, b0.v, (short)0, acc00, false, false);
        acc01 = __builtin_amdgcn_wmma_f32_16x16x32_bf16(false, a0.v, false, b1.v, (short)0, acc01, false, false);
        acc10 = __builtin_amdgcn_wmma_f32_16x16x32_bf16(false, a1.v, false, b0.v, (short)0, acc10, false, false);
        acc11 = __builtin_amdgcn_wmma_f32_16x16x32_bf16(false, a1.v, false, b1.v, (short)0, acc11, false, false);
    }

#pragma unroll
    for (int t = 0; t < 2; ++t) {
        if ((N % 32 == 0) || (ncol[t] < N)) {
            const float badd = bias ? bias[ncol[t]] : 0.0f;
            const v8f* ac0 = t ? &acc01 : &acc00;
            const v8f* ac1 = t ? &acc11 : &acc10;
#pragma unroll
            for (int r = 0; r < 8; ++r) {
                const int row0 = m0 + r + half * 8;
                C[(long)row0 * N + ncol[t]]        = (*ac0)[r] + badd;
                C[(long)(row0 + 16) * N + ncol[t]] = (*ac1)[r] + badd;
            }
        }
    }
}

// ---------------- f32 -> bf16 bulk convert (weights, 4-wide) ----------------
__global__ void cvt_bf16_kernel(const float* __restrict__ in,
                                __bf16* __restrict__ out, long n4)
{
    const long i = (long)blockIdx.x * blockDim.x + threadIdx.x;
    if (i >= n4) return;
    const float4 v = *(const float4*)(in + i * 4);
    union { __bf16 h[4]; uint2 q; } p;
    p.h[0] = (__bf16)v.x; p.h[1] = (__bf16)v.y;
    p.h[2] = (__bf16)v.z; p.h[3] = (__bf16)v.w;
    *(uint2*)(out + i * 4) = p.q;
}

// =====================================================================
// FFT filter: per (b,d) sequence, radix-2 FFT in LDS, complex filter,
// inverse FFT, + residual x.  'ortho' fwd*inv == 1/N total.
// =====================================================================
#define LFFT 2048
#define LOG2L 11

static __device__ void fft_stages(float* re, float* im, int tid, int nt, float dir)
{
    for (int s = 1; s <= LOG2L; ++s) {
        const int m = 1 << s, half = m >> 1;
        for (int idx = tid; idx < (LFFT >> 1); idx += nt) {
            const int g = idx / half;
            const int j = idx - g * half;
            const int i0 = g * m + j, i1 = i0 + half;
            const float ang = dir * (-6.28318530717958647692f) * (float)j / (float)m;
            const float wr = __cosf(ang), wi = __sinf(ang);
            const float xr = re[i1], xi = im[i1];
            const float tr = xr * wr - xi * wi;
            const float ti = xr * wi + xi * wr;
            const float ur = re[i0], ui = im[i0];
            re[i1] = ur - tr; im[i1] = ui - ti;
            re[i0] = ur + tr; im[i0] = ui + ti;
        }
        __syncthreads();
    }
}

__global__ void fft_filter_kernel(const float* __restrict__ x,
                                  const float* __restrict__ filt,  // (L/2+1, D, 2)
                                  float* __restrict__ out)
{
    __shared__ float re[LFFT];
    __shared__ float im[LFFT];
    const int b = blockIdx.x / D_;
    const int d = blockIdx.x % D_;
    const int tid = threadIdx.x;
    const int nt  = blockDim.x;

    for (int i = tid; i < LFFT; i += nt) {
        const unsigned r = __brev((unsigned)i) >> (32 - LOG2L);
        re[r] = x[((long)(b * L_ + i)) * D_ + d];
        im[r] = 0.0f;
    }
    __syncthreads();
    fft_stages(re, im, tid, nt, 1.0f);

    for (int k = tid; k < LFFT; k += nt) {
        float wr, wi;
        if (k <= LFFT / 2) {
            wr = filt[((long)k * D_ + d) * 2 + 0];
            wi = filt[((long)k * D_ + d) * 2 + 1];
        } else {
            const int kk = LFFT - k;
            wr =  filt[((long)kk * D_ + d) * 2 + 0];
            wi = -filt[((long)kk * D_ + d) * 2 + 1];
        }
        const float a = re[k], bI = im[k];
        re[k] = a * wr - bI * wi;
        im[k] = a * wi + bI * wr;
    }
    __syncthreads();

    for (int i = tid; i < LFFT; i += nt) {
        const unsigned r = __brev((unsigned)i) >> (32 - LOG2L);
        if (i < (int)r) {
            float t = re[i]; re[i] = re[r]; re[r] = t;
            t = im[i]; im[i] = im[r]; im[r] = t;
        }
    }
    __syncthreads();
    fft_stages(re, im, tid, nt, -1.0f);

    const float inv = 1.0f / (float)LFFT;
    for (int i = tid; i < LFFT; i += nt) {
        const long idx = ((long)(b * L_ + i)) * D_ + d;
        out[idx] = re[i] * inv + x[idx];
    }
}

// =====================================================================
// LayerNorm over D=128 (one wave per row), optional residual, optional
// flipped read of `in`, optional extra bf16 copy of the output.
// =====================================================================
__global__ void add_ln_kernel(const float* __restrict__ in,
                              const float* __restrict__ res,
                              const float* __restrict__ w,
                              const float* __restrict__ bvec,
                              float* __restrict__ out,
                              __bf16* __restrict__ out_bf,
                              int rows, float eps, int flip)
{
    const int wave = threadIdx.x >> 5, lane = threadIdx.x & 31;
    const int row = blockIdx.x * (blockDim.x >> 5) + wave;
    if (row >= rows) return;
    int in_row = row;
    if (flip) {
        const int b = row / L_, l = row % L_;
        in_row = b * L_ + (L_ - 1 - l);
    }
    const int c = lane * 4;
    float4 v = *(const float4*)(in + (long)in_row * D_ + c);
    if (res) {
        const float4 rr = *(const float4*)(res + (long)row * D_ + c);
        v.x += rr.x; v.y += rr.y; v.z += rr.z; v.w += rr.w;
    }
    float s = v.x + v.y + v.z + v.w;
    for (int off = 16; off; off >>= 1) s += __shfl_xor(s, off);
    const float mean = s * (1.0f / D_);
    float vs = (v.x-mean)*(v.x-mean) + (v.y-mean)*(v.y-mean)
             + (v.z-mean)*(v.z-mean) + (v.w-mean)*(v.w-mean);
    for (int off = 16; off; off >>= 1) vs += __shfl_xor(vs, off);
    const float inv = rsqrtf(vs * (1.0f / D_) + eps);
    const float4 wv = *(const float4*)(w + c);
    const float4 bv = *(const float4*)(bvec + c);
    float4 o;
    o.x = (v.x-mean)*inv*wv.x + bv.x;
    o.y = (v.y-mean)*inv*wv.y + bv.y;
    o.z = (v.z-mean)*inv*wv.z + bv.z;
    o.w = (v.w-mean)*inv*wv.w + bv.w;
    *(float4*)(out + (long)row * D_ + c) = o;
    if (out_bf) {
        union { __bf16 h[4]; uint2 q; } p;
        p.h[0] = (__bf16)o.x; p.h[1] = (__bf16)o.y;
        p.h[2] = (__bf16)o.z; p.h[3] = (__bf16)o.w;
        *(uint2*)(out_bf + (long)row * D_ + c) = p.q;
    }
}

// =====================================================================
// Depthwise causal conv (k=4) + bias + silu; writes u f32 (scan) + bf16 (GEMM)
// =====================================================================
__global__ void conv_silu_kernel(const float* __restrict__ xz,
                                 const float* __restrict__ cw,   // (256,4)
                                 const float* __restrict__ cb,   // (256)
                                 float* __restrict__ u,
                                 __bf16* __restrict__ ubf)
{
    const long i = (long)blockIdx.x * blockDim.x + threadIdx.x;
    const long total = (long)B_ * L_ * DI_;
    if (i >= total) return;
    const int  d  = (int)(i % DI_);
    const long bl = i / DI_;
    const int  l  = (int)(bl % L_);
    const long b  = bl / L_;
    const float4 wv = *(const float4*)(cw + d * 4);
    float acc = cb[d];
    if (l >= 1) acc += wv.z * xz[((b * L_ + (l-1)) * (2*DI_)) + d];
    if (l >= 2) acc += wv.y * xz[((b * L_ + (l-2)) * (2*DI_)) + d];
    if (l >= 3) acc += wv.x * xz[((b * L_ + (l-3)) * (2*DI_)) + d];
    acc += wv.w * xz[((b * L_ + l) * (2*DI_)) + d];
    const float o = acc * sigmoidf_(acc);
    u[i] = o;
    ubf[i] = (__bf16)o;
}

// =====================================================================
// Selective scan, fully fused: dt-projection (K=8 dot), softplus+bias,
// h-recurrence, y = sum_s h*C (shuffle tree), + u*D, * silu(z).
// 16 lanes = 16 states of one (b,d); 2 channels/wave.  y out in bf16
// (only consumed by the out-proj WMMA GEMM).
// =====================================================================
__global__ void scan_kernel(const float* __restrict__ u,      // (B,L,256) f32
                            const float* __restrict__ bc,     // (B,L,40): dt|B|C
                            const float* __restrict__ xz,     // (B,L,512): z at +256
                            const float* __restrict__ dt_w,   // (256,8)
                            const float* __restrict__ A_log,  // (256,16)
                            const float* __restrict__ dt_b,   // (256)
                            const float* __restrict__ Dp,     // (256)
                            __bf16* __restrict__ ybf)         // (B,L,256) bf16
{
    const int wave = threadIdx.x >> 5, lane = threadIdx.x & 31;
    const int s = lane & 15, halfw = lane >> 4;
    const int pid = (blockIdx.x * (blockDim.x >> 5) + wave) * 2 + halfw;
    const int b = pid / DI_, d = pid % DI_;
    if (b >= B_) return;

    const float4 w0 = *(const float4*)(dt_w + d * DTR_);
    const float4 w1 = *(const float4*)(dt_w + d * DTR_ + 4);
    const float Av  = -__expf(A_log[d * DS_ + s]);
    const float dtb = dt_b[d];
    const float Dd  = Dp[d];
    float h = 0.0f;
    const long base = (long)b * L_;
    for (int l = 0; l < L_; ++l) {
        const long row = base + l;
        const float4 q0 = *(const float4*)(bc + row * XPN_);
        const float4 q1 = *(const float4*)(bc + row * XPN_ + 4);
        float dp = dtb + q0.x*w0.x + q0.y*w0.y + q0.z*w0.z + q0.w*w0.w
                       + q1.x*w1.x + q1.y*w1.y + q1.z*w1.z + q1.w*w1.w;
        const float dl = (dp > 20.0f) ? dp : log1pf(__expf(dp));
        const float uv = u[row * DI_ + d];
        const float Bv = bc[row * XPN_ + DTR_ + s];
        const float Cv = bc[row * XPN_ + DTR_ + DS_ + s];
        h = __expf(dl * Av) * h + dl * Bv * uv;
        float part = h * Cv;
        part += __shfl_xor(part, 1);
        part += __shfl_xor(part, 2);
        part += __shfl_xor(part, 4);
        part += __shfl_xor(part, 8);
        if (s == 0) {
            const float z = xz[row * (2 * DI_) + DI_ + d];
            ybf[row * DI_ + d] = (__bf16)((part + uv * Dd) * (z * sigmoidf_(z)));
        }
    }
}

// ---------------- small elementwise helpers ----------------
__global__ void flip_kernel(const float* __restrict__ in, __bf16* __restrict__ out)
{
    const long i = (long)blockIdx.x * blockDim.x + threadIdx.x;
    const long total = (long)B_ * L_ * D_;
    if (i >= total) return;
    const int c = (int)(i % D_);
    const long bl = i / D_;
    const int l = (int)(bl % L_);
    const long b = bl / L_;
    out[i] = (__bf16)in[(b * L_ + (L_ - 1 - l)) * D_ + c];
}

__global__ void add2_kernel(const float* __restrict__ a, const float* __restrict__ b,
                            float* __restrict__ out, __bf16* __restrict__ out_bf, long n)
{
    const long i = (long)blockIdx.x * blockDim.x + threadIdx.x;
    if (i >= n) return;
    const float v = a[i] + b[i];
    out[i] = v;
    out_bf[i] = (__bf16)v;
}

__global__ void glu_gate_kernel(const float* __restrict__ t,   // (B,L,256) f32
                                __bf16* __restrict__ gv)       // (B,L,128) bf16
{
    const long i = (long)blockIdx.x * blockDim.x + threadIdx.x;
    const long total = (long)B_ * L_ * D_;
    if (i >= total) return;
    const int c = (int)(i % D_);
    const long row = i / D_;
    const float v = t[row * (2 * D_) + c];
    const float g = t[row * (2 * D_) + D_ + c];
    gv[i] = (__bf16)(v * sigmoidf_(g));
}

// =====================================================================
// Host launcher
// =====================================================================
template<int N, int K>
static void launch_gemm(const __bf16* A, const __bf16* W, const float* bias,
                        float* C, hipStream_t stream)
{
    dim3 grid(MROWS / 128, (N + 31) / 32);   // 32x32 macro-tile/wave, 4 waves/block
    gemm_wmma_kernel<N, K><<<grid, 128, 0, stream>>>(A, W, bias, C);
}

static void cvt_bf16(const float* in, __bf16* out, long n, hipStream_t stream)
{
    const long n4 = n / 4;
    cvt_bf16_kernel<<<(int)((n4 + 255) / 256), 256, 0, stream>>>(in, out, n4);
}

extern "C" void kernel_launch(void* const* d_in, const int* in_sizes, int n_in,
                              void* d_out, int out_size, void* d_ws, size_t ws_size,
                              hipStream_t stream)
{
    (void)in_sizes; (void)n_in; (void)out_size; (void)ws_size;
    const float* x         = (const float*)d_in[0];
    const float* filt_w    = (const float*)d_in[1];
    const float* filt_ln_w = (const float*)d_in[2];
    const float* filt_ln_b = (const float*)d_in[3];
    const float* fw[11];
    const float* bw[11];
    for (int i = 0; i < 11; ++i) { fw[i] = (const float*)d_in[4 + i]; bw[i] = (const float*)d_in[15 + i]; }
    const float* fc1_w  = (const float*)d_in[26];
    const float* fc1_b  = (const float*)d_in[27];
    const float* fc2_w  = (const float*)d_in[28];
    const float* fc2_b  = (const float*)d_in[29];
    const float* gln_w  = (const float*)d_in[30];
    const float* gln_b  = (const float*)d_in[31];
    float* outp = (float*)d_out;

    // ---------------- workspace carve-up ----------------
    float* ws = (float*)d_ws;
    const long BLD = (long)B_ * L_ * D_;          // 8,388,608
    float* bs_buf = ws;                           // BLD
    float* h      = ws + 1 * BLD;                 // BLD
    float* fs     = ws + 2 * BLD;                 // BLD
    float* tmp    = ws + 3 * BLD;                 // BLD
    float* xz     = ws + 4 * BLD;                 // 4*BLD (B,L,512)
    float* u      = ws + 8 * BLD;                 // 2*BLD (B,L,256); GLU t reuse
    float* bc     = ws + 10 * BLD;                // B*L*40

    __bf16* bfbase  = (__bf16*)(ws + 11 * BLD);
    __bf16* h_bf    = bfbase;                     // BLD
    __bf16* hflipbf = bfbase + 1 * BLD;           // BLD
    __bf16* u_bf    = bfbase + 2 * BLD;           // 2*BLD
    __bf16* y_bf    = bfbase + 4 * BLD;           // 2*BLD
    __bf16* gv_bf   = bfbase + 6 * BLD;           // BLD
    __bf16* wtab    = bfbase + 7 * BLD;           // bf16 weights (~0.5M)

    // bf16 weight table offsets (whole arrays, both layers)
    __bf16* w_in[2]  = { wtab,            wtab + 131072 };          // (2,512,128) f/b
    __bf16* w_xp[2]  = { wtab + 262144,   wtab + 282624 };          // (2,40,256)  f/b
    __bf16* w_out[2] = { wtab + 303104,   wtab + 368640 };          // (2,128,256) f/b
    __bf16* w_fc1    = wtab + 434176;                               // (256,128)
    __bf16* w_fc2    = wtab + 466944;                               // (128,128)

    cvt_bf16(fw[0], w_in[0],  2L*512*128, stream);
    cvt_bf16(bw[0], w_in[1],  2L*512*128, stream);
    cvt_bf16(fw[3], w_xp[0],  2L*40*256,  stream);
    cvt_bf16(bw[3], w_xp[1],  2L*40*256,  stream);
    cvt_bf16(fw[8], w_out[0], 2L*128*256, stream);
    cvt_bf16(bw[8], w_out[1], 2L*128*256, stream);
    cvt_bf16(fc1_w, w_fc1,    256L*128,   stream);
    cvt_bf16(fc2_w, w_fc2,    128L*128,   stream);

    const int LN_BLOCKS = MROWS / 8;
    auto add_ln = [&](const float* in, const float* res, const float* w_,
                      const float* b_, float* o, __bf16* obf, float eps, int flip) {
        add_ln_kernel<<<LN_BLOCKS, 256, 0, stream>>>(in, res, w_, b_, o, obf,
                                                     MROWS, eps, flip);
    };
    auto mamba = [&](const __bf16* hin_bf, const float* const* wp,
                     const __bf16* winb, const __bf16* wxpb, const __bf16* woutb,
                     int layer, float* outbuf) {
        const float* conv_w  = wp[1] + (long)layer * DI_ * 4;
        const float* conv_b  = wp[2] + (long)layer * DI_;
        const float* dt_w    = wp[4] + (long)layer * DI_ * DTR_;
        const float* dt_bv   = wp[5] + (long)layer * DI_;
        const float* A_log   = wp[6] + (long)layer * DI_ * DS_;
        const float* Dp      = wp[7] + (long)layer * DI_;
        launch_gemm<2 * DI_, D_>(hin_bf, winb + (long)layer * (2*DI_) * D_,
                                 nullptr, xz, stream);                       // (B,L,512)
        conv_silu_kernel<<<(B_ * L_ * DI_) / 256, 256, 0, stream>>>(xz, conv_w, conv_b, u, u_bf);
        launch_gemm<XPN_, DI_>(u_bf, wxpb + (long)layer * XPN_ * DI_,
                               nullptr, bc, stream);                         // (B,L,40)
        scan_kernel<<<(B_ * DI_) / 16, 256, 0, stream>>>(u, bc, xz, dt_w,
                                                         A_log, dt_bv, Dp, y_bf);
        launch_gemm<D_, DI_>(y_bf, woutb + (long)layer * D_ * DI_,
                             nullptr, outbuf, stream);                       // (B,L,128)
    };

    // 1) FFT filter + residual, then LN -> h (f32 + bf16)
    fft_filter_kernel<<<B_ * D_, 256, 0, stream>>>(x, filt_w, tmp);
    add_ln(tmp, nullptr, filt_ln_w, filt_ln_b, h, h_bf, 1e-12f, 0);

    // 2) BiMamba layers
    for (int layer = 0; layer < 2; ++layer) {
        mamba(h_bf, fw, w_in[0], w_xp[0], w_out[0], layer, tmp);
        add_ln(tmp, h, fw[9] + (long)layer * D_, fw[10] + (long)layer * D_,
               fs, nullptr, 1e-5f, 0);
        flip_kernel<<<(int)(BLD / 256), 256, 0, stream>>>(h, hflipbf);
        mamba(hflipbf, bw, w_in[1], w_xp[1], w_out[1], layer, tmp);
        add_ln(tmp, h, bw[9] + (long)layer * D_, bw[10] + (long)layer * D_,
               bs_buf, nullptr, 1e-5f, 1);
        add2_kernel<<<(int)(BLD / 256), 256, 0, stream>>>(fs, bs_buf, h, h_bf, BLD);
    }

    // 3) GLU head + final LN
    launch_gemm<2 * D_, D_>(h_bf, w_fc1, fc1_b, u, stream);          // t = (B,L,256)
    glu_gate_kernel<<<(int)(BLD / 256), 256, 0, stream>>>(u, gv_bf); // v*sig(g) -> bf16
    launch_gemm<D_, D_>(gv_bf, w_fc2, fc2_b, tmp, stream);           // gv
    add_ln(tmp, h, gln_w, gln_b, outp, nullptr, 1e-12f, 0);
}
